// BiMamba4TS_28647431864939
// MI455X (gfx1250) — matrix operand order repaired and checked by reading, not compiled, not run on hardware
//
#include <hip/hip_runtime.h>
#include <math.h>

// ---------------------------------------------------------------------------
// BiMamba4TS fused pipeline for gfx1250 (MI455X), wave32 + v_wmma_f32_16x16x32_f16
// ---------------------------------------------------------------------------

typedef __attribute__((ext_vector_type(16))) _Float16 v16h;
typedef __attribute__((ext_vector_type(8)))  _Float16 v8h;
typedef __attribute__((ext_vector_type(8)))  float    v8f;

#define THRESH_F 0.6f

// silu via fast hardware reciprocal (v_rcp_f32): result feeds fp16 anyway.
__device__ __forceinline__ float silu_f(float x) {
    return x * __builtin_amdgcn_rcpf(1.0f + __expf(-x));
}

__device__ __forceinline__ v16h make_frag(v8h lo, v8h hi) {
    v16h a;
#pragma unroll
    for (int i = 0; i < 8; ++i) { a[i] = lo[i]; a[i + 8] = hi[i]; }
    return a;
}

// A-fragment (16x32 f16, ISA 7.12.2) from LDS: lane&15 = row; per-lane 16 halfs
// = two contiguous 8-half chunks at K = klo and K = klo+16 (klo folds (lane>>4)*8).
__device__ __forceinline__ v16h load_A(const _Float16* rowptr, int klo) {
    v8h lo = *(const v8h*)(rowptr + klo);
    v8h hi = *(const v8h*)(rowptr + klo + 16);
    return make_frag(lo, hi);
}

// A-fragment gathered straight from global fp32 (each x element is consumed by
// exactly one lane), converted to fp16 in registers: no LDS round-trip needed.
__device__ __forceinline__ v16h load_A_g(const float* p, int klo) {
    const float4 u0 = *(const float4*)(p + klo);
    const float4 u1 = *(const float4*)(p + klo + 4);
    const float4 u2 = *(const float4*)(p + klo + 16);
    const float4 u3 = *(const float4*)(p + klo + 20);
    v16h a;
    a[0]  = (_Float16)u0.x; a[1]  = (_Float16)u0.y;
    a[2]  = (_Float16)u0.z; a[3]  = (_Float16)u0.w;
    a[4]  = (_Float16)u1.x; a[5]  = (_Float16)u1.y;
    a[6]  = (_Float16)u1.z; a[7]  = (_Float16)u1.w;
    a[8]  = (_Float16)u2.x; a[9]  = (_Float16)u2.y;
    a[10] = (_Float16)u2.z; a[11] = (_Float16)u2.w;
    a[12] = (_Float16)u3.x; a[13] = (_Float16)u3.y;
    a[14] = (_Float16)u3.z; a[15] = (_Float16)u3.w;
    return a;
}

// B-fragment: prep_kernel pre-swizzles weights so each lane's 16 halfs are
// contiguous (32 B) in memory: one clean global load pair per fragment.
__device__ __forceinline__ v16h load_B(const _Float16* p) {
    v8h lo = *(const v8h*)(p);
    v8h hi = *(const v8h*)(p + 8);
    return make_frag(lo, hi);
}

// ---------------------------------------------------------------------------
// Kernel 1: SRA decider. corr: [8][64][1024] -> strat[8] in {0,1}
// ---------------------------------------------------------------------------
__global__ void decide_kernel(const float* __restrict__ corr, int* __restrict__ strat) {
    const int b = blockIdx.x;
    const int tid = threadIdx.x;
    const float* cb = corr + (size_t)b * 64 * 1024;
    __shared__ float s_mean[64];
    __shared__ float s_inv[64];
    __shared__ int c_thr, c_pos;
    if (tid == 0) { c_thr = 0; c_pos = 0; }
    if (tid < 64) {
        const float* row = cb + tid * 1024;
        float s = 0.f, sq = 0.f;
        for (int i = 0; i < 1024; ++i) { float v = row[i]; s += v; sq += v * v; }
        float m = s * (1.0f / 1024.0f);
        float var = (sq - 1024.0f * m * m) * (1.0f / 1023.0f);   // ddof=1
        s_mean[tid] = m;
        s_inv[tid] = rsqrtf(fmaxf(var, 1e-30f));
    }
    __syncthreads();
    for (int p = tid; p < 4096; p += 256) {
        int s0 = p >> 6, t0 = p & 63;
        if (s0 == t0) continue;                                   // diag zeroed
        const float* r0 = cb + s0 * 1024;
        const float* r1 = cb + t0 * 1024;
        float dot = 0.f;
        for (int i = 0; i < 1024; ++i) dot += r0[i] * r1[i];
        float cv = (dot - 1024.0f * s_mean[s0] * s_mean[t0]) *
                   s_inv[s0] * s_inv[t0] * (1.0f / 1024.0f);
        if (cv > THRESH_F) atomicAdd(&c_thr, 1);
        if (cv > 0.0f)     atomicAdd(&c_pos, 1);
    }
    __syncthreads();
    if (tid == 0) {
        float ratio = (c_pos > 0) ? ((float)c_thr / (float)c_pos) : 0.0f;
        strat[b] = (ratio >= (1.0f - THRESH_F)) ? 1 : 0;
    }
}

// ---------------------------------------------------------------------------
// Kernel 2: weight prep (fp16 WMMA-fragment swizzled weights + folded tail).
// ---------------------------------------------------------------------------
__global__ void prep_kernel(const float* __restrict__ W1f, const float* __restrict__ Cwf,
                            const float* __restrict__ W1b, const float* __restrict__ Cwb,
                            const float* __restrict__ W2f, const float* __restrict__ b2f,
                            const float* __restrict__ W2b, const float* __restrict__ b2b,
                            const float* __restrict__ Wr,  const float* __restrict__ br,
                            _Float16* __restrict__ w1s, _Float16* __restrict__ cws,
                            float* __restrict__ vfvb, float* __restrict__ cscalar) {
    const int tid = threadIdx.x;
    if (blockIdx.x == 0) {
        float accf = 0.f, accb = 0.f;
        for (int g = 0; g < 256; ++g) {
            float w = Wr[g];
            accf += W2f[tid * 256 + g] * w;
            accb += W2b[tid * 256 + g] * w;
        }
        vfvb[tid] = accf;
        vfvb[256 + tid] = accb;
        if (tid == 0) {
            float c = br[0];
            for (int f = 0; f < 256; ++f) c += (b2f[f] + b2b[f]) * Wr[f];
            *cscalar = c;
        }
    }
    const int total_w1 = 2 * 32768;
    const int total_cw = 2 * 196608;
    for (int idx = blockIdx.x * blockDim.x + tid; idx < total_w1 + total_cw;
         idx += gridDim.x * blockDim.x) {
        if (idx < total_w1) {
            int d = idx >> 15;
            int r = idx & 32767;
            int h = r & 15, lane = (r >> 4) & 31, ks = (r >> 9) & 3, nt = r >> 11;
            int k = ks * 32 + ((lane >> 4) & 1) * 8 + (h & 7) + (h >> 3) * 16;
            int n = nt * 16 + (lane & 15);
            const float* W1 = d ? W1b : W1f;
            w1s[idx] = (_Float16)W1[k * 256 + n];
        } else {
            int j = idx - total_w1;
            int d = (j >= 196608) ? 1 : 0;
            int r = d ? (j - 196608) : j;
            int h = r & 15, lane = (r >> 4) & 31, kt = (r >> 9) & 7,
                nt = (r >> 12) & 15, k = r >> 16;
            int fi = kt * 32 + ((lane >> 4) & 1) * 8 + (h & 7) + (h >> 3) * 16;
            int fo = nt * 16 + (lane & 15);
            const float* Cw = d ? Cwb : Cwf;
            cws[j] = (_Float16)Cw[(fo * 256 + fi) * 3 + k];
        }
    }
}

// ---------------------------------------------------------------------------
// Kernel 3: fused per-sequence BiMamba. One 8-wave block per (b,n,s).
//  waves 0-3: forward (seq s, M-tile = wave&3); waves 4-7: backward (seq 63-s).
// ---------------------------------------------------------------------------
__launch_bounds__(256)
__global__ void bimamba_fused(const float* __restrict__ x,
                              const int* __restrict__ strat,
                              const _Float16* __restrict__ w1s,
                              const _Float16* __restrict__ cws,
                              const float* __restrict__ b1f, const float* __restrict__ b1b,
                              const float* __restrict__ cbf, const float* __restrict__ cbb,
                              const float* __restrict__ vfvb,
                              const float* __restrict__ cptr,
                              float* __restrict__ out) {
    extern __shared__ __align__(16) char smem[];
    _Float16* Hf = (_Float16*)smem;            // 66 x 264 fp16 hidden (rows 64,65 zero)
    _Float16* Hb = Hf + 66 * 264;              // 66 x 264
    float*    oacc = (float*)(Hb + 66 * 264);  // 64 floats

    const int tid  = threadIdx.x;
    const int lane = tid & 31;
    const int wave = tid >> 5;

    const int idx = blockIdx.x;
    const int s  = idx & 63;
    const int n  = (idx >> 6) & 3;
    const int b  = idx >> 8;
    const int bn = b * 4 + n;
    const int st = strat[b];
    const int s2 = 63 - s;

    if (tid < 64) oacc[tid] = 0.0f;
    for (int i = tid; i < 2 * 264; i += 256) {          // zero pad rows 64,65
        Hf[64 * 264 + i] = (_Float16)0.0f;
        Hb[64 * 264 + i] = (_Float16)0.0f;
    }

    const int d  = wave >> 2;                  // 0 = fwd, 1 = bwd
    const int mt = wave & 3;                   // M-tile (16 rows of lp)
    _Float16* H = d ? Hb : Hf;
    const _Float16* w1 = w1s + d * 32768;
    const float* b1 = d ? b1b : b1f;
    const int g8 = ((lane >> 4) & 1) * 8;      // A-fragment K sub-offset

    // ---- Phase 1+2 fused: A gathered from global (strategy-dependent),
    //      H = silu(tok @ W1 + b1), M=64, K=128, N=256 ----
    {
        const int row = mt * 16 + (lane & 15); // lp row this lane owns
        const int q = d ? s2 : s;              // sequence index for direction
        const float* xrow = x + (size_t)bn * 64 * 8192
                          + (st ? ((size_t)row * 8192 + (size_t)q * 128)
                                : ((size_t)q * 8192 + (size_t)row * 128));
        v16h a0 = load_A_g(xrow,      g8);
        v16h a1 = load_A_g(xrow, 32 + g8);
        v16h a2 = load_A_g(xrow, 64 + g8);
        v16h a3 = load_A_g(xrow, 96 + g8);
        for (int nt = 0; nt < 16; ++nt) {
            v8f acc = {};
            const _Float16* bp = w1 + nt * 2048 + lane * 16;
            acc = __builtin_amdgcn_wmma_f32_16x16x32_f16(false, a0, false, load_B(bp       ), (short)0, acc, false, false);
            acc = __builtin_amdgcn_wmma_f32_16x16x32_f16(false, a1, false, load_B(bp +  512), (short)0, acc, false, false);
            acc = __builtin_amdgcn_wmma_f32_16x16x32_f16(false, a2, false, load_B(bp + 1024), (short)0, acc, false, false);
            acc = __builtin_amdgcn_wmma_f32_16x16x32_f16(false, a3, false, load_B(bp + 1536), (short)0, acc, false, false);
            const int ncg = nt * 16 + (lane & 15);
            const float bias = b1[ncg];
            const int mbase = mt * 16 + g8;    // C/D layout: lanes>=16 hold M=8+r
#pragma unroll
            for (int r = 0; r < 8; ++r) {
                float v = silu_f(acc[r] + bias);
                H[(mbase + r) * 264 + ncg] = (_Float16)v;
            }
        }
    }
    __syncthreads();

    // ---- Phase 3: conv1d(K=3) as WMMA GEMMs + silu + dot with v = W2@Wr.
    //      4-wide N-tile chunks: each A-fragment feeds 4 WMMAs (4x less LDS). ----
    {
        const float* v  = vfvb + d * 256;
        const float* cb = d ? cbb : cbf;
        const _Float16* cwd = cws + d * 196608;
        const int arow = mt * 16 + (lane & 15);
        float tsum[8];
#pragma unroll
        for (int r = 0; r < 8; ++r) tsum[r] = 0.0f;
        for (int ntc = 0; ntc < 4; ++ntc) {
            v8f acc0 = {}, acc1 = {}, acc2 = {}, acc3 = {};
#pragma unroll
            for (int k = 0; k < 3; ++k) {
                const _Float16* hrow = H + (arow + k) * 264;
                const _Float16* bb0  = cwd + (k * 16 + ntc * 4) * 4096 + lane * 16;
#pragma unroll
                for (int kt = 0; kt < 8; ++kt) {
                    v16h a = load_A(hrow, kt * 32 + g8);
                    const _Float16* bb = bb0 + kt * 512;
                    acc0 = __builtin_amdgcn_wmma_f32_16x16x32_f16(false, a, false, load_B(bb        ), (short)0, acc0, false, false);
                    acc1 = __builtin_amdgcn_wmma_f32_16x16x32_f16(false, a, false, load_B(bb +  4096), (short)0, acc1, false, false);
                    acc2 = __builtin_amdgcn_wmma_f32_16x16x32_f16(false, a, false, load_B(bb +  8192), (short)0, acc2, false, false);
                    acc3 = __builtin_amdgcn_wmma_f32_16x16x32_f16(false, a, false, load_B(bb + 12288), (short)0, acc3, false, false);
                }
            }
#pragma unroll
            for (int j = 0; j < 4; ++j) {
                v8f acc = (j == 0) ? acc0 : (j == 1) ? acc1 : (j == 2) ? acc2 : acc3;
                const int fo = (ntc * 4 + j) * 16 + (lane & 15);
                const float bias = cb[fo];
                const float w = v[fo];
#pragma unroll
                for (int r = 0; r < 8; ++r) tsum[r] += silu_f(acc[r] + bias) * w;
            }
        }
        // reduce over the 16 N-lanes of each half-wave (xor of bits 0..3 only)
#pragma unroll
        for (int r = 0; r < 8; ++r) {
            float t = tsum[r];
            t += __shfl_xor(t, 1, 32);
            t += __shfl_xor(t, 2, 32);
            t += __shfl_xor(t, 4, 32);
            t += __shfl_xor(t, 8, 32);
            tsum[r] = t;
        }
        if ((lane & 15) == 0) {
            const int mbase = mt * 16 + g8;
#pragma unroll
            for (int r = 0; r < 8; ++r) atomicAdd(&oacc[mbase + r], tsum[r]);
        }
    }
    __syncthreads();

    // ---- Phase 4: out[b,n,s,l] = fwd+bwd + ((b2f+b2b)@Wr + br) ----
    if (tid < 62) {
        out[((size_t)bn * 64 + s) * 62 + tid] = oacc[tid] + cptr[0];
    }
}

// ---------------------------------------------------------------------------
extern "C" void kernel_launch(void* const* d_in, const int* in_sizes, int n_in,
                              void* d_out, int out_size, void* d_ws, size_t ws_size,
                              hipStream_t stream) {
    (void)in_sizes; (void)n_in; (void)out_size; (void)ws_size;
    const float* x    = (const float*)d_in[0];
    const float* corr = (const float*)d_in[1];
    const float* W1f  = (const float*)d_in[2];
    const float* b1f  = (const float*)d_in[3];
    const float* Cwf  = (const float*)d_in[4];
    const float* Cbf  = (const float*)d_in[5];
    const float* W2f  = (const float*)d_in[6];
    const float* b2f  = (const float*)d_in[7];
    const float* W1b  = (const float*)d_in[8];
    const float* b1b  = (const float*)d_in[9];
    const float* Cwb  = (const float*)d_in[10];
    const float* Cbb  = (const float*)d_in[11];
    const float* W2b  = (const float*)d_in[12];
    const float* b2b  = (const float*)d_in[13];
    const float* Wr   = (const float*)d_in[14];
    const float* br   = (const float*)d_in[15];
    float* out = (float*)d_out;

    char* ws = (char*)d_ws;
    int*      strat   = (int*)ws;                          // [0, 64)
    float*    vfvb    = (float*)(ws + 64);                 // [64, 2112): vf, vb
    float*    cscalar = (float*)(ws + 2112);               // [2112, 2176)
    _Float16* w1s     = (_Float16*)(ws + 2176);            // 2*32768 halfs
    _Float16* cws     = (_Float16*)(ws + 2176 + 131072);   // 2*196608 halfs

    decide_kernel<<<8, 256, 0, stream>>>(corr, strat);
    prep_kernel<<<512, 256, 0, stream>>>(W1f, Cwf, W1b, Cwb, W2f, b2f, W2b, b2b,
                                         Wr, br, w1s, cws, vfvb, cscalar);
    const size_t smem = (size_t)(2 * 66 * 264) * sizeof(_Float16)
                      + 64 * sizeof(float);                // 69952 B < 320 KB
    bimamba_fused<<<2048, 256, smem, stream>>>(x, strat, w1s, cws,
                                               b1f, b1b, Cbf, Cbb,
                                               vfvb, cscalar, out);
}
